// HybirdPhopGNN_53017076301936
// MI455X (gfx1250) — compile-verified, matrix operand order checked by reading, not run on hardware
//
#include <hip/hip_runtime.h>
#include <math.h>

// ---------------- static problem config ----------------
#define GB   64          // graphs
#define NPG  256         // nodes per graph
#define NT   16384       // total nodes
#define INCH 128
#define HID  256
#define LL   3
#define PP   3
#define KF   8

typedef __bf16 bf16_t;
typedef __attribute__((ext_vector_type(16))) __bf16 v16bf;
typedef __attribute__((ext_vector_type(8)))  __bf16 v8bf;
typedef __attribute__((ext_vector_type(8)))  float  v8f;
typedef __attribute__((ext_vector_type(2)))  float  v2f;

// =====================================================================
// Generic batched bf16 WMMA GEMM:  C[b] = A[b] (MxK, row major, lda)
//                                        * BT[b]^T (BT is N x K row major, ldb)
// flags bit0: C += (load existing C) ; bit1: init with bias[n]
// Block: 128 threads = 4 waves; block tile 64x128; wave tile 32x64
// (2 A-frags x 4 B-frags = 8 WMMAs per k-step, 12 b128 loads).
// Requires M%64==0, N%128==0, K%32==0 (true for all call sites).
// =====================================================================
__global__ void gemm_bf16(const bf16_t* __restrict__ A,
                          const bf16_t* __restrict__ BT,
                          const float*  __restrict__ bias,
                          float* __restrict__ C,
                          int K, int lda, int ldb, int ldc,
                          long long sA, long long sB, long long sC,
                          int flags)
{
    const int wave = threadIdx.x >> 5;
    const int lane = threadIdx.x & 31;
    const int h = lane >> 4;          // half (0/1)
    const int r = lane & 15;
    const long long zb = blockIdx.z;
    A  += zb * sA;
    BT += zb * sB;
    C  += zb * sC;
    const int m0 = blockIdx.y * 64 + (wave & 1) * 32;   // 2 row-frags: m0, m0+16
    const int n0 = blockIdx.x * 128 + (wave >> 1) * 64; // 4 col-frags of 16

    v8f acc[2][4];
#pragma unroll
    for (int wt = 0; wt < 2; ++wt)
#pragma unroll
        for (int t = 0; t < 4; ++t) {
            float binit = (flags & 2) ? bias[n0 + t * 16 + r] : 0.0f;
#pragma unroll
            for (int v = 0; v < 8; ++v) acc[wt][t][v] = binit;
            if (flags & 1) {
#pragma unroll
                for (int v = 0; v < 8; ++v)
                    acc[wt][t][v] += C[(long long)(m0 + wt * 16 + 8 * h + v) * ldc
                                       + n0 + t * 16 + r];
            }
        }

    for (int k0 = 0; k0 < K; k0 += 32) {
        v16bf afr[2];
#pragma unroll
        for (int wt = 0; wt < 2; ++wt) {
            const bf16_t* ap = A + (long long)(m0 + wt * 16 + r) * lda + k0 + h * 8;
            if (k0 + 32 < K) __builtin_prefetch(ap + 32, 0, 1);
            v8bf alo = *(const v8bf*)ap;
            v8bf ahi = *(const v8bf*)(ap + 16);
#pragma unroll
            for (int i = 0; i < 8; ++i) { afr[wt][i] = alo[i]; afr[wt][8 + i] = ahi[i]; }
        }
#pragma unroll
        for (int t = 0; t < 4; ++t) {
            const bf16_t* bp = BT + (long long)(n0 + t * 16 + r) * ldb + k0 + h * 8;
            v8bf blo = *(const v8bf*)bp;
            v8bf bhi = *(const v8bf*)(bp + 16);
            v16bf bfr;
#pragma unroll
            for (int i = 0; i < 8; ++i) { bfr[i] = blo[i]; bfr[8 + i] = bhi[i]; }
            acc[0][t] = __builtin_amdgcn_wmma_f32_16x16x32_bf16(
                false, afr[0], false, bfr, (short)0, acc[0][t], false, false);
            acc[1][t] = __builtin_amdgcn_wmma_f32_16x16x32_bf16(
                false, afr[1], false, bfr, (short)0, acc[1][t], false, false);
        }
    }

#pragma unroll
    for (int wt = 0; wt < 2; ++wt)
#pragma unroll
        for (int t = 0; t < 4; ++t)
#pragma unroll
            for (int v = 0; v < 8; ++v)
                C[(long long)(m0 + wt * 16 + 8 * h + v) * ldc + n0 + t * 16 + r]
                    = acc[wt][t][v];
}

// =====================================================================
// f32 WMMA (16x16x4) batched symmetric GEMM: Sim[b] = Xn[b] * Xn[b]^T
// (exact f32 path: topology decision is precision sensitive)
// =====================================================================
__global__ void simgemm_f32(const float* __restrict__ Xn, float* __restrict__ Sim)
{
    const int wave = threadIdx.x >> 5;
    const int lane = threadIdx.x & 31;
    const int h = lane >> 4;
    const int r = lane & 15;
    const long long zb = blockIdx.z;
    const float* X = Xn + zb * (long long)(NPG * HID);
    float* C = Sim + zb * (long long)(NPG * NPG);
    const int m0 = blockIdx.y * 64 + wave * 16;
    const int n0 = blockIdx.x * 64;

    v8f acc[4] = {};
    for (int k0 = 0; k0 < HID; k0 += 4) {
        v2f a = *(const v2f*)(X + (long long)(m0 + r) * HID + k0 + 2 * h);
#pragma unroll
        for (int t = 0; t < 4; ++t) {
            v2f b = *(const v2f*)(X + (long long)(n0 + t * 16 + r) * HID + k0 + 2 * h);
            acc[t] = __builtin_amdgcn_wmma_f32_16x16x4_f32(
                false, a, false, b, (short)0, acc[t], false, false);
        }
    }
#pragma unroll
    for (int t = 0; t < 4; ++t)
#pragma unroll
        for (int v = 0; v < 8; ++v)
            C[(long long)(m0 + 8 * h + v) * NPG + n0 + t * 16 + r] = acc[t][v];
}

// ---------------- elementwise / helper kernels ----------------

// f32 -> bf16 convert (grid-stride)
__global__ void k_cvt(const float* __restrict__ s, bf16_t* __restrict__ d, long long n)
{
    long long i = (long long)blockIdx.x * blockDim.x + threadIdx.x;
    long long stride = (long long)gridDim.x * blockDim.x;
    for (; i < n; i += stride) d[i] = (bf16_t)s[i];
}

// batched transpose + convert: dst[n*ldd + m] = (bf16) src[m*lds + n]
__global__ void k_tcvt(const float* __restrict__ src, bf16_t* __restrict__ dst,
                       int M, int N, int lds, int ldd, long long sS, long long sD)
{
    src += (long long)blockIdx.z * sS;
    dst += (long long)blockIdx.z * sD;
    int idx = blockIdx.x * 256 + threadIdx.x;
    if (idx < M * N) {
        int m = idx / N, n = idx % N;
        dst[(long long)n * ldd + m] = (bf16_t)src[(long long)m * lds + n];
    }
}

// scatter adjacency (A[g,dst,src]=1, AT[g,src,dst]=1) + self loops
__global__ void k_scatter(const int* __restrict__ ei, int E,
                          unsigned short* __restrict__ A, unsigned short* __restrict__ AT)
{
    int t = blockIdx.x * 256 + threadIdx.x;
    const unsigned short one = 0x3F80; // bf16 1.0
    if (t < E) {
        int s = ei[t], d = ei[E + t];
        int g = s >> 8, sl = s & 255, dl = d & 255;
        A [((long long)g * NPG + dl) * NPG + sl] = one;
        AT[((long long)g * NPG + sl) * NPG + dl] = one;
    } else {
        int u = t - E;
        if (u < NT) {
            int g = u >> 8, n = u & 255;
            A [((long long)g * NPG + n) * NPG + n] = one;
            AT[((long long)g * NPG + n) * NPG + n] = one;
        }
    }
}

// per-row L2 normalization (one wave per node row)
__global__ void k_rownorm(const float* __restrict__ X, float* __restrict__ Xn)
{
    int row = blockIdx.x, lane = threadIdx.x;
    const float* xr = X + (long long)row * HID;
    float v[8]; float s = 0.f;
#pragma unroll
    for (int t = 0; t < 8; ++t) { v[t] = xr[lane + 32 * t]; s += v[t] * v[t]; }
#pragma unroll
    for (int m = 16; m >= 1; m >>= 1) s += __shfl_xor(s, m, 32);
    float inv = 1.0f / (sqrtf(s) + 1e-12f);
    float* o = Xn + (long long)row * HID;
#pragma unroll
    for (int t = 0; t < 8; ++t) o[lane + 32 * t] = v[t] * inv;
}

// top-K (K=8) smallest similarities per row -> farthest graph FA / FA^T (+diag)
__global__ void k_topk(const float* __restrict__ sim,
                       unsigned short* __restrict__ FA, unsigned short* __restrict__ FAT)
{
    int row = blockIdx.x * 256 + threadIdx.x; // 0..NT-1
    int g = row >> 8, i = row & 255;
    const float* sr = sim + (long long)row * NPG;
    const unsigned short one = 0x3F80;
    float lastv = -1e30f; int lasti = -1;
    for (int k = 0; k < KF; ++k) {
        float bestv = 1e30f; int besti = NPG;
        for (int j = 0; j < NPG; ++j) {
            float v = sr[j];
            bool gt = (v > lastv) || (v == lastv && j > lasti);
            bool lt = (v < bestv) || (v == bestv && j < besti);
            if (gt && lt) { bestv = v; besti = j; }
        }
        FA [(long long)row * NPG + besti] = one;
        FAT[((long long)g * NPG + besti) * NPG + i] = one;
        lastv = bestv; lasti = besti;
    }
    FA [(long long)row * NPG + i] = one;
    FAT[((long long)g * NPG + i) * NPG + i] = one;
}

// U_p = {A-I, A2-A, A3-A2}, pre-scaled by 1/deg (deg = row sum |U_p|) -> bf16
// one wave per (graph,row)
__global__ void k_makeU(const bf16_t* __restrict__ Ab, const float* __restrict__ M2,
                        const float* __restrict__ M3, bf16_t* __restrict__ Un)
{
    int row = blockIdx.x, lane = threadIdx.x;
    int i = row & 255;
    const long long base = (long long)row * NPG;
    float u[3][8]; float s0 = 0.f, s1 = 0.f, s2 = 0.f;
#pragma unroll
    for (int t = 0; t < 8; ++t) {
        int j = lane + 32 * t;
        float a  = (float)Ab[base + j];
        float m2 = M2[base + j];
        float m3 = M3[base + j];
        float u1 = a - (j == i ? 1.0f : 0.0f);
        float u2 = m2 - a;
        float u3 = m3 - m2;
        u[0][t] = u1; u[1][t] = u2; u[2][t] = u3;
        s0 += fabsf(u1); s1 += fabsf(u2); s2 += fabsf(u3);
    }
#pragma unroll
    for (int m = 16; m >= 1; m >>= 1) {
        s0 += __shfl_xor(s0, m, 32);
        s1 += __shfl_xor(s1, m, 32);
        s2 += __shfl_xor(s2, m, 32);
    }
    const long long pstride = (long long)GB * NPG * NPG;
    float inv[3];
    inv[0] = s0 > 0.f ? 1.0f / s0 : 0.0f;
    inv[1] = s1 > 0.f ? 1.0f / s1 : 0.0f;
    inv[2] = s2 > 0.f ? 1.0f / s2 : 0.0f;
#pragma unroll
    for (int p = 0; p < 3; ++p)
#pragma unroll
        for (int t = 0; t < 8; ++t) {
            int j = lane + 32 * t;
            Un[p * pstride + base + j] = (bf16_t)(u[p][t] * inv[p]);
        }
}

// GraphNorm + leaky ReLU; per-graph per-channel stats; writes f32 and bf16.
// grid 64 blocks, 256 threads (thread = channel). in-place f32 safe.
__global__ void k_graphnorm(const float* __restrict__ in,
                            const float* __restrict__ gamma, const float* __restrict__ beta,
                            const float* __restrict__ alpha,
                            float* __restrict__ outf, bf16_t* __restrict__ outb)
{
    int g = blockIdx.x, c = threadIdx.x;
    const float* p = in + (long long)g * NPG * HID;
    float s = 0.f;
    for (int j = 0; j < NPG; ++j) s += p[j * HID + c];
    float mean = s * (1.0f / NPG);
    float am = alpha[c] * mean;
    float s2 = 0.f;
    for (int j = 0; j < NPG; ++j) { float d = p[j * HID + c] - am; s2 += d * d; }
    float inv = rsqrtf(s2 * (1.0f / NPG) + 1e-5f);
    float gg = gamma[c], bb = beta[c];
    long long obase = (long long)g * NPG * HID;
    for (int j = 0; j < NPG; ++j) {
        float d = (p[j * HID + c] - am) * inv * gg + bb;
        d = d > 0.f ? d : 0.01f * d;
        outf[obase + j * HID + c] = d;
        outb[obase + j * HID + c] = (bf16_t)d;
    }
}

// x = sigmoid(gp)*cx + (1-sigmoid(gp))*fx + x   (elementwise, in-place on x)
__global__ void k_combine(const float* __restrict__ gp, const float* __restrict__ cx,
                          const float* __restrict__ fx, float* __restrict__ x, long long n)
{
    long long i = (long long)blockIdx.x * blockDim.x + threadIdx.x;
    long long stride = (long long)gridDim.x * blockDim.x;
    for (; i < n; i += stride) {
        float gv = 1.0f / (1.0f + expf(-gp[i]));
        x[i] = gv * cx[i] + (1.0f - gv) * fx[i] + x[i];
    }
}

// global mean pool per graph
__global__ void k_meanpool(const float* __restrict__ x, float* __restrict__ out)
{
    int g = blockIdx.x, c = threadIdx.x;
    float s = 0.f;
    for (int j = 0; j < NPG; ++j) s += x[((long long)g * NPG + j) * HID + c];
    out[g * HID + c] = s * (1.0f / NPG);
}

// =====================================================================
extern "C" void kernel_launch(void* const* d_in, const int* in_sizes, int n_in,
                              void* d_out, int out_size, void* d_ws, size_t ws_size,
                              hipStream_t stream)
{
    (void)n_in; (void)out_size; (void)ws_size;
    const float* in_x  = (const float*)d_in[0];
    const int*   ei    = (const int*)d_in[1];
    const float* Wemb  = (const float*)d_in[3];
    const float* bemb  = (const float*)d_in[4];
    const float* cW    = (const float*)d_in[5];
    const float* cb    = (const float*)d_in[6];
    const float* fW    = (const float*)d_in[7];
    const float* fb    = (const float*)d_in[8];
    const float* gG    = (const float*)d_in[9];
    const float* gBt   = (const float*)d_in[10];
    const float* gA    = (const float*)d_in[11];
    const float* fG    = (const float*)d_in[12];
    const float* fBt   = (const float*)d_in[13];
    const float* fA    = (const float*)d_in[14];
    const float* gateW = (const float*)d_in[15];
    const float* gateb = (const float*)d_in[16];
    const int E = in_sizes[1] / 2;

    // ---------------- workspace layout ----------------
    const size_t SZ_XF    = (size_t)NT * HID * 4;            // 16.8 MB
    const size_t SZ_UN    = (size_t)PP * GB * NPG * NPG * 2; // 25.2 MB
    const size_t SZ_WEMBT = (size_t)HID * INCH * 2;
    const size_t SZ_CWT   = (size_t)LL * HID * (PP * HID) * 2;
    const size_t SZ_GWT   = (size_t)HID * HID * 2;

    char* w = (char*)d_ws;
    size_t off = 0;
    auto take = [&](size_t bytes) -> char* {
        char* p = w + off;
        off = (off + bytes + 255) & ~(size_t)255;
        return p;
    };
    float*  x     = (float*)take(SZ_XF);
    bf16_t* SUn   = (bf16_t*)take(SZ_UN);
    bf16_t* FUn   = (bf16_t*)take(SZ_UN);
    bf16_t* WembT = (bf16_t*)take(SZ_WEMBT);
    bf16_t* cWT   = (bf16_t*)take(SZ_CWT);
    bf16_t* fWT   = (bf16_t*)take(SZ_CWT);
    bf16_t* gWT1  = (bf16_t*)take(SZ_GWT);
    bf16_t* gWT2  = (bf16_t*)take(SZ_GWT);
    char* scratch = w + off;

    // setup-phase views into scratch
    bf16_t* Abf = (bf16_t*)(scratch + 0);           //  8.39 MB
    bf16_t* ATb = (bf16_t*)(scratch + 8388608);     //  8.39 MB
    float*  M2  = (float*) (scratch + 16777216);    // 16.78 MB
    bf16_t* M2b = (bf16_t*)(scratch + 33554432);    //  8.39 MB
    float*  M3  = (float*) (scratch + 41943040);    // 16.78 MB
    float*  xn  = (float*) (scratch + 58720256);    // 16.78 MB
    float*  sim = (float*) (scratch + 75497472);    // 16.78 MB
    bf16_t* x0b = (bf16_t*)(scratch + 92274688);    //  4.19 MB

    // layer-phase views (alias, used after setup done)
    float*  cx   = (float*) (scratch + 0);
    float*  fx   = (float*) (scratch + 16777216);
    bf16_t* cxb  = (bf16_t*)(scratch + 33554432);
    bf16_t* fxb  = (bf16_t*)(scratch + 41943040);
    float*  gate = (float*) (scratch + 50331648);
    bf16_t* xTb  = (bf16_t*)(scratch + 67108864);
    float*  agg  = (float*) (scratch + 75497472);   // 50.33 MB (NT x 768)
    bf16_t* aggb = (bf16_t*)(scratch + 125829120);  // 25.17 MB

    const long long G2 = (long long)NPG * NPG;       // 65536
    const long long PSTRIDE = (long long)GB * G2;    // plane stride in Un

    // ---------------- weight transposes (f32 -> bf16, BT layout) ----------------
    k_tcvt<<<dim3(128, 1, 1), 256, 0, stream>>>(Wemb, WembT, INCH, HID, HID, INCH, 0, 0);
    for (int l = 0; l < LL; ++l)
        for (int p = 0; p < PP; ++p) {
            k_tcvt<<<dim3(256, 1, 1), 256, 0, stream>>>(
                cW + ((long long)(l * PP + p) * HID) * HID,
                cWT + (long long)l * HID * (PP * HID) + p * HID,
                HID, HID, HID, PP * HID, 0, 0);
            k_tcvt<<<dim3(256, 1, 1), 256, 0, stream>>>(
                fW + ((long long)(l * PP + p) * HID) * HID,
                fWT + (long long)l * HID * (PP * HID) + p * HID,
                HID, HID, HID, PP * HID, 0, 0);
        }
    k_tcvt<<<dim3(256, 1, 1), 256, 0, stream>>>(gateW, gWT1, HID, HID, HID, HID, 0, 0);
    k_tcvt<<<dim3(256, 1, 1), 256, 0, stream>>>(gateW + (long long)HID * HID, gWT2, HID, HID, HID, HID, 0, 0);

    // ---------------- embedding: x = in_x @ Wemb + bemb ----------------
    k_cvt<<<4096, 256, 0, stream>>>(in_x, x0b, (long long)NT * INCH);
    gemm_bf16<<<dim3(HID / 128, NT / 64, 1), 128, 0, stream>>>(
        x0b, WembT, bemb, x, INCH, INCH, INCH, HID, 0, 0, 0, /*flags=*/2);

    // ---------------- structural adjacency & U hops ----------------
    hipMemsetAsync(Abf, 0, (size_t)GB * G2 * 2, stream);
    hipMemsetAsync(ATb, 0, (size_t)GB * G2 * 2, stream);
    k_scatter<<<(E + NT + 255) / 256, 256, 0, stream>>>(ei, E, (unsigned short*)Abf, (unsigned short*)ATb);
    // A2 = A @ A   (BT = A^T)
    gemm_bf16<<<dim3(2, 4, GB), 128, 0, stream>>>(Abf, ATb, nullptr, M2,
        NPG, NPG, NPG, NPG, G2, G2, G2, 0);
    k_cvt<<<4096, 256, 0, stream>>>(M2, M2b, (long long)GB * G2);
    // A3 = A2 @ A
    gemm_bf16<<<dim3(2, 4, GB), 128, 0, stream>>>(M2b, ATb, nullptr, M3,
        NPG, NPG, NPG, NPG, G2, G2, G2, 0);
    k_makeU<<<NT, 32, 0, stream>>>(Abf, M2, M3, SUn);

    // ---------------- feature (farthest) graph & U hops ----------------
    k_rownorm<<<NT, 32, 0, stream>>>(x, xn);
    simgemm_f32<<<dim3(4, 4, GB), 128, 0, stream>>>(xn, sim);
    hipMemsetAsync(Abf, 0, (size_t)GB * G2 * 2, stream);   // reuse as FA
    hipMemsetAsync(ATb, 0, (size_t)GB * G2 * 2, stream);   // reuse as FA^T
    k_topk<<<NT / 256, 256, 0, stream>>>(sim, (unsigned short*)Abf, (unsigned short*)ATb);
    gemm_bf16<<<dim3(2, 4, GB), 128, 0, stream>>>(Abf, ATb, nullptr, M2,
        NPG, NPG, NPG, NPG, G2, G2, G2, 0);
    k_cvt<<<4096, 256, 0, stream>>>(M2, M2b, (long long)GB * G2);
    gemm_bf16<<<dim3(2, 4, GB), 128, 0, stream>>>(M2b, ATb, nullptr, M3,
        NPG, NPG, NPG, NPG, G2, G2, G2, 0);
    k_makeU<<<NT, 32, 0, stream>>>(Abf, M2, M3, FUn);

    // ---------------- layers ----------------
    const long long nElem = (long long)NT * HID;
    for (int l = 0; l < LL; ++l) {
        // --- structural P-hop conv: cx = sum_p (Un_p x) Wc_p + cb ---
        k_tcvt<<<dim3(256, 1, GB), 256, 0, stream>>>(x, xTb, NPG, HID, HID, NPG, G2, G2);
        for (int p = 0; p < PP; ++p)
            gemm_bf16<<<dim3(2, 4, GB), 128, 0, stream>>>(
                SUn + p * PSTRIDE, xTb, nullptr, agg + p * HID,
                NPG, NPG, NPG, PP * HID, G2, G2, (long long)NPG * PP * HID, 0);
        k_cvt<<<4096, 256, 0, stream>>>(agg, aggb, (long long)NT * PP * HID);
        gemm_bf16<<<dim3(HID / 128, NT / 64, 1), 128, 0, stream>>>(
            aggb, cWT + (long long)l * HID * (PP * HID), cb + l * HID, cx,
            PP * HID, PP * HID, PP * HID, HID, 0, 0, 0, 2);
        k_graphnorm<<<GB, HID, 0, stream>>>(cx, gG + l * HID, gBt + l * HID, gA + l * HID, cx, cxb);

        // --- feature P-hop conv on cx: fx = sum_p (FUn_p cx) Wf_p + fb ---
        k_tcvt<<<dim3(256, 1, GB), 256, 0, stream>>>(cx, xTb, NPG, HID, HID, NPG, G2, G2);
        for (int p = 0; p < PP; ++p)
            gemm_bf16<<<dim3(2, 4, GB), 128, 0, stream>>>(
                FUn + p * PSTRIDE, xTb, nullptr, agg + p * HID,
                NPG, NPG, NPG, PP * HID, G2, G2, (long long)NPG * PP * HID, 0);
        k_cvt<<<4096, 256, 0, stream>>>(agg, aggb, (long long)NT * PP * HID);
        gemm_bf16<<<dim3(HID / 128, NT / 64, 1), 128, 0, stream>>>(
            aggb, fWT + (long long)l * HID * (PP * HID), fb + l * HID, fx,
            PP * HID, PP * HID, PP * HID, HID, 0, 0, 0, 2);
        k_graphnorm<<<GB, HID, 0, stream>>>(fx, fG + l * HID, fBt + l * HID, fA + l * HID, fx, fxb);

        // --- gate = sigmoid(cx @ gW1 + fx @ gW2 + gb); x = g*cx+(1-g)*fx+x ---
        gemm_bf16<<<dim3(HID / 128, NT / 64, 1), 128, 0, stream>>>(
            cxb, gWT1, gateb, gate, HID, HID, HID, HID, 0, 0, 0, 2);
        gemm_bf16<<<dim3(HID / 128, NT / 64, 1), 128, 0, stream>>>(
            fxb, gWT2, nullptr, gate, HID, HID, HID, HID, 0, 0, 0, 1);
        k_combine<<<4096, 256, 0, stream>>>(gate, cx, fx, x, nElem);
    }

    // ---------------- global mean pool ----------------
    k_meanpool<<<GB, HID, 0, stream>>>(x, (float*)d_out);
}
